// BitLinear_37778532335922
// MI455X (gfx1250) — 8-bit matrix operand order repaired and checked by reading, compiled, not run on hardware
//
#include <hip/hip_runtime.h>
#include <stdint.h>

typedef __attribute__((ext_vector_type(8))) int v8i;

#define D_IN   1024
#define D_OUT  1024
#define KSTEPS (D_IN / 64)

__global__ void pack_weights_kernel(const int* __restrict__ w,
                                    uint8_t* __restrict__ wfrag) {
  int g = blockIdx.x * blockDim.x + threadIdx.x;
  int4 w4 = reinterpret_cast<const int4*>(w)[g];
  int n = g >> 8;
  int k = (g << 2) & (D_IN - 1);
  int ntile = n >> 4, nrow = n & 15;
  int kstep = k >> 6, koff = k & 63;
  int q2 = koff >> 5;
  int h  = (koff >> 4) & 1;
  int b  = koff & 15;
  int lane = nrow + (h << 4);
  size_t off = ((size_t)(ntile * KSTEPS + kstep) << 11) + (lane << 5) + (q2 << 4) + b;
  uint32_t pack = (uint32_t)(w4.x & 255)        | ((uint32_t)(w4.y & 255) << 8) |
                  ((uint32_t)(w4.z & 255) << 16) | ((uint32_t)(w4.w & 255) << 24);
  *reinterpret_cast<uint32_t*>(wfrag + off) = pack;
}

__global__ void rms_quant_kernel(const float* __restrict__ x,
                                 const float* __restrict__ wscale_p,
                                 uint8_t* __restrict__ qfrag,
                                 float* __restrict__ dq) {
  int m = blockIdx.x;
  int t = threadIdx.x;
  float4 v = reinterpret_cast<const float4*>(x + (size_t)m * D_IN)[t];
  float ss = v.x * v.x + v.y * v.y + v.z * v.z + v.w * v.w;
  float am = fmaxf(fmaxf(fabsf(v.x), fabsf(v.y)), fmaxf(fabsf(v.z), fabsf(v.w)));
  #pragma unroll
  for (int o = 16; o > 0; o >>= 1) {
    ss += __shfl_xor(ss, o, 32);
    am = fmaxf(am, __shfl_xor(am, o, 32));
  }
  __shared__ float s_ss[8];
  __shared__ float s_am[8];
  int wave = t >> 5, lane = t & 31;
  if (lane == 0) { s_ss[wave] = ss; s_am[wave] = am; }
  __syncthreads();
  float tss = 0.0f, tam = 0.0f;
  #pragma unroll
  for (int i = 0; i < 8; ++i) {
    tss += s_ss[i];
    tam = fmaxf(tam, s_am[i]);
  }
  float r   = rsqrtf(tss * (1.0f / (float)D_IN) + 1e-6f);
  float den = fmaxf(tam * r, 1e-5f);
  float s   = 127.0f / den;
  float rs  = r * s;

  if (t == 0) dq[m] = (*wscale_p) * den * (1.0f / 127.0f);

  int q0 = (int)rintf(v.x * rs);
  int q1 = (int)rintf(v.y * rs);
  int q2v = (int)rintf(v.z * rs);
  int q3 = (int)rintf(v.w * rs);
  q0 = q0 < -128 ? -128 : (q0 > 127 ? 127 : q0);
  q1 = q1 < -128 ? -128 : (q1 > 127 ? 127 : q1);
  q2v = q2v < -128 ? -128 : (q2v > 127 ? 127 : q2v);
  q3 = q3 < -128 ? -128 : (q3 > 127 ? 127 : q3);
  uint32_t pack = (uint32_t)(q0 & 255)        | ((uint32_t)(q1 & 255) << 8) |
                  ((uint32_t)(q2v & 255) << 16) | ((uint32_t)(q3 & 255) << 24);

  int k = t << 2;
  int mtile = m >> 4, mrow = m & 15;
  int kstep = k >> 6, koff = k & 63;
  int b = koff & 15;
  int h = (koff >> 4) & 1;
  int c = koff >> 5;
  int lane32 = mrow + (h << 4);
  size_t off = ((size_t)(mtile * KSTEPS + kstep) << 11) + (lane32 << 5) + (c << 4) + b;
  *reinterpret_cast<uint32_t*>(qfrag + off) = pack;
}

__device__ __forceinline__ v8i load_frag(const uint8_t* p) {
  int4 lo = *reinterpret_cast<const int4*>(p);
  int4 hi = *reinterpret_cast<const int4*>(p + 16);
  v8i r;
  r[0] = lo.x; r[1] = lo.y; r[2] = lo.z; r[3] = lo.w;
  r[4] = hi.x; r[5] = hi.y; r[6] = hi.z; r[7] = hi.w;
  return r;
}

__global__ void __launch_bounds__(256)
gemm_iu8_kernel(const uint8_t* __restrict__ qfrag,
                const uint8_t* __restrict__ wfrag,
                const float* __restrict__ dq,
                const float* __restrict__ bias,
                float* __restrict__ out) {
  int lane   = threadIdx.x & 31;
  int wave   = threadIdx.x >> 5;
  int mtile  = blockIdx.x * 8 + wave;
  int ntile0 = blockIdx.y * 8;

  const uint8_t* ApBase = qfrag + (((size_t)mtile * KSTEPS) << 11) + (lane << 5);
  const uint8_t* BpBase = wfrag + (((size_t)ntile0 * KSTEPS) << 11) + (lane << 5);

  v8i acc[8] = {};

  v8i aCur = load_frag(ApBase);
  v8i bCur = load_frag(BpBase);

  for (int ks = 0; ks < KSTEPS; ++ks) {
    int ks1 = (ks + 1) & (KSTEPS - 1);
    #pragma unroll
    for (int j = 0; j < 8; ++j) {
      int jn  = (j + 1) & 7;
      int ksn = (j == 7) ? ks1 : ks;
      v8i bNext = load_frag(BpBase + ((size_t)(jn * KSTEPS + ksn) << 11));
      v8i aNext;
      if (j == 7) aNext = load_frag(ApBase + ((size_t)ks1 << 11));
      acc[j] = __builtin_amdgcn_wmma_i32_16x16x64_iu8(
           true, aCur,  true, bCur, acc[j],
           false,  false);
      bCur = bNext;
      if (j == 7) aCur = aNext;
    }
  }

  int rowbase = (mtile << 4) + ((lane >> 4) << 3);
  float4 d0 = *reinterpret_cast<const float4*>(dq + rowbase);
  float4 d1 = *reinterpret_cast<const float4*>(dq + rowbase + 4);
  float dqv[8] = {d0.x, d0.y, d0.z, d0.w, d1.x, d1.y, d1.z, d1.w};
  int coll = lane & 15;

  #pragma unroll
  for (int j = 0; j < 8; ++j) {
    int colg = ((ntile0 + j) << 4) + coll;
    float bs = bias[colg];
    #pragma unroll
    for (int v = 0; v < 8; ++v) {
      out[((size_t)(rowbase + v) << 10) + colg] = (float)acc[j][v] * dqv[v] + bs;
    }
  }
}

extern "C" void kernel_launch(void* const* d_in, const int* in_sizes, int n_in,
                              void* d_out, int out_size, void* d_ws, size_t ws_size,
                              hipStream_t stream) {
  const float* x      = (const float*)d_in[0];
  const int*   w      = (const int*)d_in[1];
  const float* wscale = (const float*)d_in[2];
  const float* bias   = (const float*)d_in[3];
  float*       out    = (float*)d_out;

  int M = in_sizes[0] / D_IN;

  uint8_t* ws    = (uint8_t*)d_ws;
  uint8_t* qfrag = ws;
  uint8_t* wfrag = ws + (size_t)M * D_IN;
  float*   dq    = (float*)(ws + (size_t)M * D_IN + (size_t)D_OUT * D_IN);

  pack_weights_kernel<<<(D_OUT * D_IN / 4) / 256, 256, 0, stream>>>(w, wfrag);

  rms_quant_kernel<<<M, 256, 0, stream>>>(x, wscale, qfrag, dq);

  dim3 grid(M / 128, D_OUT / 128);
  gemm_iu8_kernel<<<grid, 256, 0, stream>>>(qfrag, wfrag, dq, bias, out);
}
